// GeometricAttention_42245298323805
// MI455X (gfx1250) — matrix operand order repaired and checked by reading, not compiled, not run on hardware
//
#include <hip/hip_runtime.h>
#include <hip/hip_bf16.h>

typedef __attribute__((ext_vector_type(16))) _Float16 v16h;
typedef __attribute__((ext_vector_type(8)))  _Float16 v8h;
typedef __attribute__((ext_vector_type(8)))  float    v8f;

// Fixed problem shape: B=2, T=2048, C=1024, H=8, HD=128
#define TT 2048
#define CC 1024

// ---------------------------------------------------------------------------
// f32 -> f16 convert
// ---------------------------------------------------------------------------
__global__ void cvt_f16_kernel(const float* __restrict__ src,
                               _Float16* __restrict__ dst, int n) {
  int i = blockIdx.x * blockDim.x + threadIdx.x;
  if (i < n) dst[i] = (_Float16)src[i];
}

// ---------------------------------------------------------------------------
// Build folded q/k weights: Weqk [64][1024] f16
// rows 0..31  : q-ext, h-major, 4 per head: [scale*Wq[h*3+d] (d<3), dir_h . Wq rows]
// rows 32..63 : k-ext:                       [Wk[h*3+d]       (d<3), dir_h . Wk rows]
// dirs[h] = (s0,s1,s2)/sqrt(3), s0=+1 iff h&4, s1=+1 iff h&2, s2=+1 iff h&1
// ---------------------------------------------------------------------------
__global__ void build_weqk_kernel(const float* __restrict__ Wq,
                                  const float* __restrict__ Wk,
                                  const float* __restrict__ scale,
                                  _Float16* __restrict__ Weqk) {
  int idx = blockIdx.x * blockDim.x + threadIdx.x;  // 64*1024
  if (idx >= 64 * CC) return;
  int r = idx >> 10, c = idx & (CC - 1);
  int isK = r >> 5;
  int h = (r >> 2) & 7, d = r & 3;
  const float* W = isK ? Wk : Wq;
  float val;
  if (d < 3) {
    val = W[(h * 3 + d) * CC + c] * (isK ? 1.0f : scale[0]);
  } else {
    const float r3 = 0.57735026918962576f;  // 1/sqrt(3)
    float s0 = (h & 4) ? r3 : -r3;
    float s1 = (h & 2) ? r3 : -r3;
    float s2 = (h & 1) ? r3 : -r3;
    val = s0 * W[(h * 3 + 0) * CC + c] +
          s1 * W[(h * 3 + 1) * CC + c] +
          s2 * W[(h * 3 + 2) * CC + c];
  }
  Weqk[(size_t)r * CC + c] = (_Float16)val;
}

// ---------------------------------------------------------------------------
// WMMA fragment loaders (wave32, v_wmma_f32_16x16x32_f16 layouts per CDNA5 ISA)
//  A (16x32, M x K): lane row M = lane&15; elem i<8 -> K = hi*8+i,
//                    elem i>=8 -> K = 16+hi*8+(i-8)  (two contiguous 16B runs)
//  B (32x16, K x N): lane col N = lane&15; same K order as A (two 16B runs)
//  C/D f32 (16x16):  elem r -> (M = r+8*hi, N = lane&15)
// ---------------------------------------------------------------------------
__device__ __forceinline__ v16h load_afrag(const _Float16* rowptr, int k0, int hi) {
  v16h a;
  *(v8h*)&a       = *(const v8h*)(rowptr + k0 + hi * 8);
  *((v8h*)&a + 1) = *(const v8h*)(rowptr + k0 + 16 + hi * 8);
  return a;
}
__device__ __forceinline__ v16h load_bfrag(const _Float16* rowptr, int k0, int hi) {
  v16h b;
  *(v8h*)&b       = *(const v8h*)(rowptr + k0 + hi * 8);
  *((v8h*)&b + 1) = *(const v8h*)(rowptr + k0 + 16 + hi * 8);
  return b;
}

// ---------------------------------------------------------------------------
// WMMA GEMM:  Cmn = sum_k A[m][k] * Bw[n][k]   (i.e. A @ Bw^T), K = 1024.
// A: f16 [M][1024] row-major, Bw: f16 [N][1024] row-major.
// One wave per block: 16 rows x (NT*16) cols. Double-buffered k-loop; the
// sched_barrier pins next-stage loads ahead of current-stage WMMAs.
// __launch_bounds__(32,2): relax VGPR budget so the double buffer stays
// in registers (no scratch spills) while keeping 2 waves/EU.
// MODE 0: f32 row-major output, leading dim LDC.
// MODE 1: f16 output into Vt[b][h][d][t]: col n -> (h=n>>7, d=n&127),
//         row m -> (b=m>>11, t=m&2047); 8 packed halfs contiguous in t.
// ---------------------------------------------------------------------------
template <int NT, int MODE, int LDC>
__global__ __launch_bounds__(32, 2)
void gemm_wmma_kernel(const _Float16* __restrict__ A,
                      const _Float16* __restrict__ Bw,
                      void* __restrict__ Cout) {
  constexpr int K = CC;  // 1024
  const int lane = threadIdx.x & 31;
  const int l15 = lane & 15, hi = lane >> 4;
  const int mbase = blockIdx.x * 16;
  const int nbase = blockIdx.y * (NT * 16);

  v8f acc[NT];
#pragma unroll
  for (int n = 0; n < NT; ++n) acc[n] = (v8f){};

  const _Float16* arow = A + (size_t)(mbase + l15) * K;
  const _Float16* brow[NT];
#pragma unroll
  for (int n = 0; n < NT; ++n)
    brow[n] = Bw + (size_t)(nbase + n * 16 + l15) * K;

  // prologue: stage k0 = 0
  v16h a0 = load_afrag(arow, 0, hi);
  v16h b0[NT];
#pragma unroll
  for (int n = 0; n < NT; ++n) b0[n] = load_bfrag(brow[n], 0, hi);

#pragma unroll 4
  for (int k0 = 0; k0 < K - 32; k0 += 32) {
    // issue next-stage loads first; barrier keeps them ahead of the WMMAs
    v16h a1 = load_afrag(arow, k0 + 32, hi);
    v16h b1[NT];
#pragma unroll
    for (int n = 0; n < NT; ++n) b1[n] = load_bfrag(brow[n], k0 + 32, hi);
    __builtin_amdgcn_sched_barrier(0);
#pragma unroll
    for (int n = 0; n < NT; ++n)
      acc[n] = __builtin_amdgcn_wmma_f32_16x16x32_f16(
          false, a0, false, b0[n], (short)0, acc[n], false, false);
    __builtin_amdgcn_sched_barrier(0);
    a0 = a1;
#pragma unroll
    for (int n = 0; n < NT; ++n) b0[n] = b1[n];
  }
  // epilogue stage
#pragma unroll
  for (int n = 0; n < NT; ++n)
    acc[n] = __builtin_amdgcn_wmma_f32_16x16x32_f16(
        false, a0, false, b0[n], (short)0, acc[n], false, false);

  if constexpr (MODE == 0) {
    float* C = (float*)Cout;
#pragma unroll
    for (int n = 0; n < NT; ++n)
#pragma unroll
      for (int r = 0; r < 8; ++r)
        C[(size_t)(mbase + r + 8 * hi) * LDC + nbase + n * 16 + l15] = acc[n][r];
  } else {
    _Float16* Vt = (_Float16*)Cout;
    const int b = mbase >> 11;              // batch (T=2048 rows each)
    const int t0 = (mbase & (TT - 1)) + 8 * hi;
#pragma unroll
    for (int n = 0; n < NT; ++n) {
      const int col = nbase + n * 16 + l15;  // h*128 + d
      v8h pk;
#pragma unroll
      for (int r = 0; r < 8; ++r) pk[r] = (_Float16)acc[n][r];
      *(v8h*)(Vt + (size_t)(b * 1024 + col) * TT + t0) = pk;
    }
  }
}

// ---------------------------------------------------------------------------
// Flash attention, 1 wave per (b,h, 16-query-row tile).
// qk: f32 [B*T][64]  (cols 0..31: qe h-major 4/head; cols 32..63: ke)
// Vt: f16 [B][H][128][T]
// attnh: f16 [B*T][1024]  (= [b][t][h*128+d])
// __launch_bounds__(32,2): keep bf[8]+acc[8] (~190 VGPRs) in registers.
// ---------------------------------------------------------------------------
__global__ __launch_bounds__(32, 2)
void attn_kernel(const float* __restrict__ qk,
                 const _Float16* __restrict__ Vt,
                 _Float16* __restrict__ attnh) {
  const int lane = threadIdx.x & 31;
  const int l15 = lane & 15, hi = lane >> 4;
  const int mtile = blockIdx.x;   // 0..127
  const int bh = blockIdx.y;      // 0..15
  const int b = bh >> 3, h = bh & 7;
  const int mbase = mtile * 16;
  const int m = mbase + l15;      // this lane's query row (A-operand layout)

  const float4 qv = *(const float4*)(qk + ((size_t)(b * TT + m) * 64) + h * 4);
  const float* keBase = qk + (size_t)b * TT * 64 + 32 + h * 4;
  const _Float16* vtBase = Vt + (size_t)(b * 8 + h) * 128 * TT;

  float rmax = -1e30f, rsum = 0.f;
  v8f acc[8];
#pragma unroll
  for (int n = 0; n < 8; ++n) acc[n] = (v8f){};

  const int ntiles = (mbase + 16 + 31) >> 5;  // causal: keys <= mbase+15
  for (int tile = 0; tile < ntiles; ++tile) {
    const int jbase = tile * 32;

    // ---- issue ALL V B-fragment loads first; sched_barrier pins them so the
    //      (younger) ke loads + score/softmax VALU work hides their latency.
    //      VMEM completes in order, so once scores consumed the last ke load
    //      every bf fragment has landed -> WMMA chain runs waitless. ----
    v16h bf[8];
#pragma unroll
    for (int n = 0; n < 8; ++n) {
      const _Float16* vrow =
          vtBase + (size_t)(n * 16 + l15) * TT + jbase + hi * 8;
      *(v8h*)&bf[n]       = *(const v8h*)(vrow);
      *((v8h*)&bf[n] + 1) = *(const v8h*)(vrow + 16);
    }
    __builtin_amdgcn_sched_barrier(0);

    // ---- scores for this lane's 16 K-slots (A-operand layout), 4-dim dot ----
    float s[16];
#pragma unroll
    for (int i = 0; i < 16; ++i) {
      const int kidx = jbase + ((i < 8) ? (hi * 8 + i) : (16 + hi * 8 + (i - 8)));
      float4 kv = *(const float4*)(keBase + (size_t)kidx * 64);
      float d = qv.x * kv.x + qv.y * kv.y + qv.z * kv.z + qv.w * kv.w;
      s[i] = (kidx <= m) ? d : -1e30f;  // causal mask (select, keeps EXEC full)
    }

    // ---- online softmax: row = lane&15, partner lane = lane^16 ----
    float tmax = s[0];
#pragma unroll
    for (int i = 1; i < 16; ++i) tmax = fmaxf(tmax, s[i]);
    tmax = fmaxf(tmax, __shfl_xor(tmax, 16, 32));
    const float mnew = fmaxf(rmax, tmax);
    const float corr = __expf(rmax - mnew);

    float tsum = 0.f;
    v16h pa;
#pragma unroll
    for (int i = 0; i < 16; ++i) {
      float p = __expf(s[i] - mnew);
      tsum += p;
      pa[i] = (_Float16)p;              // P tile already in A-operand layout
    }
    tsum += __shfl_xor(tsum, 16, 32);
    rsum = rsum * corr + tsum;
    rmax = mnew;

    // redistribute per-row corr to C-layout rows (row M=r+8*hi lives in lane M)
    float cr[8];
#pragma unroll
    for (int r = 0; r < 8; ++r) cr[r] = __shfl(corr, r + 8 * hi, 32);

    // ---- rescale accumulators, then 8 back-to-back WMMAs ----
#pragma unroll
    for (int n = 0; n < 8; ++n)
#pragma unroll
      for (int r = 0; r < 8; ++r) acc[n][r] *= cr[r];
#pragma unroll
    for (int n = 0; n < 8; ++n)
      acc[n] = __builtin_amdgcn_wmma_f32_16x16x32_f16(
          false, pa, false, bf[n], (short)0, acc[n], false, false);
  }

  // ---- normalize and store f16 in [b][t][h*128+d] ----
  const float inv = 1.f / rsum;
  float ir[8];
#pragma unroll
  for (int r = 0; r < 8; ++r) ir[r] = __shfl(inv, r + 8 * hi, 32);
#pragma unroll
  for (int n = 0; n < 8; ++n) {
    const int col = h * 128 + n * 16 + l15;
#pragma unroll
    for (int r = 0; r < 8; ++r) {
      const int t = mbase + r + 8 * hi;
      attnh[(size_t)(b * TT + t) * CC + col] = (_Float16)(acc[n][r] * ir[r]);
    }
  }
}

// ---------------------------------------------------------------------------
extern "C" void kernel_launch(void* const* d_in, const int* in_sizes, int n_in,
                              void* d_out, int out_size, void* d_ws, size_t ws_size,
                              hipStream_t stream) {
  (void)in_sizes; (void)n_in; (void)out_size; (void)ws_size;
  const float* x     = (const float*)d_in[0];  // [2,2048,1024]
  const float* Wq    = (const float*)d_in[1];  // [24,1024]
  const float* Wk    = (const float*)d_in[2];  // [24,1024]
  const float* Wv    = (const float*)d_in[3];  // [1024,1024]
  const float* Wo    = (const float*)d_in[4];  // [1024,1024]
  const float* scale = (const float*)d_in[5];  // [1]
  float* out = (float*)d_out;                  // [2,2048,1024]

  const int Mrows = 2 * TT;        // 4096
  const int NX = Mrows * CC;       // 4,194,304

  char* ws = (char*)d_ws;
  _Float16* xh    = (_Float16*)(ws);                  //  8,388,608 B
  _Float16* Wvh   = (_Float16*)(ws + 8388608);        //  2,097,152 B
  _Float16* Woh   = (_Float16*)(ws + 10485760);       //  2,097,152 B
  _Float16* Weqk  = (_Float16*)(ws + 12582912);       //    131,072 B
  float*    qkbuf = (float*)   (ws + 12713984);       //  1,048,576 B
  _Float16* Vt    = (_Float16*)(ws + 13762560);       //  8,388,608 B
  _Float16* attnh = xh;  // xh dead after the two x-GEMMs; reuse (stream-ordered)

  // 1) converts + folded q/k weight build
  cvt_f16_kernel<<<(NX + 255) / 256, 256, 0, stream>>>(x, xh, NX);
  cvt_f16_kernel<<<(CC * CC + 255) / 256, 256, 0, stream>>>(Wv, Wvh, CC * CC);
  cvt_f16_kernel<<<(CC * CC + 255) / 256, 256, 0, stream>>>(Wo, Woh, CC * CC);
  build_weqk_kernel<<<(64 * CC + 255) / 256, 256, 0, stream>>>(Wq, Wk, scale, Weqk);

  // 2) qk = xh @ Weqk^T  -> f32 [4096][64]
  gemm_wmma_kernel<4, 0, 64><<<dim3(Mrows / 16, 1), 32, 0, stream>>>(
      xh, Weqk, (void*)qkbuf);

  // 3) Vt = (xh @ Wvh^T) stored transposed f16 [B][H][128][T]
  gemm_wmma_kernel<4, 1, 0><<<dim3(Mrows / 16, CC / 64), 32, 0, stream>>>(
      xh, Wvh, (void*)Vt);

  // 4) flash attention -> attnh f16 [4096][1024]
  attn_kernel<<<dim3(TT / 16, 16), 32, 0, stream>>>(qkbuf, Vt, attnh);

  // 5) out = attnh @ Woh^T -> f32 [4096][1024]
  gemm_wmma_kernel<4, 0, CC><<<dim3(Mrows / 16, CC / 64), 32, 0, stream>>>(
      attnh, Woh, (void*)out);
}
